// MxIntViTSelfAttentionHead_51711406243932
// MI455X (gfx1250) — matrix operand order repaired and checked by reading, not compiled, not run on hardware
//
#include <hip/hip_runtime.h>
#include <cmath>

// MXInt ViT self-attention head, fused in LDS for MI455X (gfx1250, wave32).
// B=256, S=512, D=64.  QK^T and PV via v_wmma_f32_16x16x32_f16 (mantissas are
// integers |m|<=128 -> exact in f16; per-row V exponent folded into V before
// the contraction).  Scores never touch HBM: 232KB LDS per workgroup.

#define S_LEN    512
#define D_DIM    64
#define QT       32
#define NTHREADS 256

typedef __attribute__((ext_vector_type(16))) _Float16 v16h;
typedef __attribute__((ext_vector_type(8)))  _Float16 v8h;
typedef __attribute__((ext_vector_type(8)))  float    v8f;

// ---- LDS layout (byte offsets inside the dynamic allocation) ----
#define OFF_KS   0u        // Ks: [512][64] f16   (65536 B)  K mantissas
#define OFF_VT   65536u    // Vt: [64][512] f16   (65536 B)  V^T with 2^(ev-7) folded in
#define OFF_SC   131072u   // Sc: [32][512] f32   (65536 B)  scores -> mantissas; later ctx [32][64]
#define OFF_PS   196608u   // Ps: [32][512] f16   (32768 B)  prob mantissas
#define OFF_QS   229376u   // Qs: [32][64] f16    (4096 B)   Q mantissas
#define OFF_EK   233472u   // 512 x i32 async staging -> overwritten with f32 2^(ek-7)
#define OFF_EV   235520u   // 512 x i32 async staging (read raw in V conversion)
#define OFF_EQ   237568u   // 32  x i32 -> f32 2^(eq-7)
#define OFF_PSC  237696u   // 32  x f32 row scales 2^(e_p-7)
#define SMEM_BYTES 237824u

#define FLT_MIN_POS 1.1754943508222875e-38f   // 2^-126

// ---- CDNA5 async copy: global -> LDS, tracked with ASYNCcnt ----
__device__ __forceinline__ void async_ld_b32(unsigned ldsByteOff, const void* gaddr) {
  asm volatile("global_load_async_to_lds_b32 %0, %1, off"
               :: "v"(ldsByteOff), "v"(gaddr) : "memory");
}
__device__ __forceinline__ void wait_async0() {
  asm volatile("s_wait_asynccnt 0" ::: "memory");
}

// ---- WMMA fragment loaders (ISA 7.12.2 layouts, wave32) ----
// A (16-bit 16x32): lane m=l16 holds K {kb+0..7, kb+16..23} (lanes 0-15)
//                   or {kb+8..15, kb+24..31} (lanes 16-31), from row m.
__device__ __forceinline__ v16h fragA(const _Float16* row, int kbase, int hi) {
  union { v16h v; v8h h[2]; } u;
  const int o = kbase + (hi ? 8 : 0);
  u.h[0] = *(const v8h*)(row + o);
  u.h[1] = *(const v8h*)(row + o + 16);
  return u.v;
}
// B (16-bit 32x16): lane holds column n=l16, K {kb+0..7, kb+16..23}
// (lanes 0-15) or {kb+8..15, kb+24..31} (lanes 16-31).  With B stored as
// rows of the *transposed* operand, this is again a contiguous row read.
__device__ __forceinline__ v16h fragB(const _Float16* row, int kbase, int hi) {
  union { v16h v; v8h h[2]; } u;
  const int o = kbase + (hi ? 8 : 0);
  u.h[0] = *(const v8h*)(row + o);
  u.h[1] = *(const v8h*)(row + o + 16);
  return u.v;
}

__device__ __forceinline__ v8f wmma_f16(v16h a, v16h b, v8f c) {
  // (neg_a, A, neg_b, B, c_mod, C, reuse_a, reuse_b)
  return __builtin_amdgcn_wmma_f32_16x16x32_f16(false, a, false, b, (short)0, c,
                                                false, false);
}

// 8-lane (one MXInt row / 8 threads) reductions, wave32-safe xor masks
__device__ __forceinline__ float gmax8(float v) {
  v = fmaxf(v, __shfl_xor(v, 1, 32));
  v = fmaxf(v, __shfl_xor(v, 2, 32));
  v = fmaxf(v, __shfl_xor(v, 4, 32));
  return v;
}
__device__ __forceinline__ float gsum8(float v) {
  v += __shfl_xor(v, 1, 32);
  v += __shfl_xor(v, 2, 32);
  v += __shfl_xor(v, 4, 32);
  return v;
}

__global__ __launch_bounds__(NTHREADS, 1)
void mxint_attn_head_kernel(const float* __restrict__ mq, const int* __restrict__ eq,
                            const float* __restrict__ mk, const int* __restrict__ ek,
                            const float* __restrict__ mv, const int* __restrict__ ev,
                            float* __restrict__ out)
{
  extern __shared__ char smem[];
  _Float16* Ks = (_Float16*)(smem + OFF_KS);
  _Float16* Vt = (_Float16*)(smem + OFF_VT);
  float*    Sc = (float*)   (smem + OFF_SC);
  _Float16* Ps = (_Float16*)(smem + OFF_PS);
  _Float16* Qs = (_Float16*)(smem + OFF_QS);

  const int tid  = threadIdx.x;
  const int b    = blockIdx.y;
  const int q0   = blockIdx.x * QT;
  const int lane = tid & 31;
  const int wave = tid >> 5;
  const int l16  = lane & 15;
  const int hi   = (lane >> 4) & 1;

  const size_t batch_md = (size_t)b * S_LEN * D_DIM;
  const size_t batch_e  = (size_t)b * S_LEN;

  // ---------------- Phase 0: async exponent loads; K/Q f32->f16 ------------
  if (tid < QT) async_ld_b32(OFF_EQ + 4u*tid, eq + batch_e + q0 + tid);
  for (int i = tid; i < S_LEN; i += NTHREADS) {
    async_ld_b32(OFF_EK + 4u*i, ek + batch_e + i);
    async_ld_b32(OFF_EV + 4u*i, ev + batch_e + i);
  }
  {
    const float4* Kg = (const float4*)(mk + batch_md);
    for (int q = tid; q < S_LEN * D_DIM / 4; q += NTHREADS) {
      float4 x = Kg[q];
      _Float16* d = Ks + q * 4;
      d[0] = (_Float16)x.x; d[1] = (_Float16)x.y;
      d[2] = (_Float16)x.z; d[3] = (_Float16)x.w;
    }
    const float4* Qg = (const float4*)(mq + batch_md + (size_t)q0 * D_DIM);
    for (int q = tid; q < QT * D_DIM / 4; q += NTHREADS) {
      float4 x = Qg[q];
      _Float16* d = Qs + q * 4;
      d[0] = (_Float16)x.x; d[1] = (_Float16)x.y;
      d[2] = (_Float16)x.z; d[3] = (_Float16)x.w;
    }
  }
  wait_async0();
  __syncthreads();

  // ------- Phase 1: exponent -> scale (in place); V -> V^T with ev folded --
  {
    volatile int* eqI = (volatile int*)(smem + OFF_EQ);
    float*        eqs = (float*)(smem + OFF_EQ);
    if (tid < QT) { int e = eqI[tid]; eqs[tid] = exp2f((float)e - 7.f); }
    volatile int* ekI = (volatile int*)(smem + OFF_EK);
    float*        eks = (float*)(smem + OFF_EK);
    for (int i = tid; i < S_LEN; i += NTHREADS) {
      int e = ekI[i]; eks[i] = exp2f((float)e - 7.f);
    }
    volatile int* evI = (volatile int*)(smem + OFF_EV);
    const float4* Vg  = (const float4*)(mv + batch_md);
    for (int q = tid; q < S_LEN * D_DIM / 4; q += NTHREADS) {
      const int t  = q >> 4;            // 16 float4 per 64-wide row
      const int d0 = (q & 15) * 4;
      const float sc = exp2f((float)evI[t] - 7.f);   // exact: |mv|<2^7 -> f16 exact
      float4 x = Vg[q];
      Vt[(d0 + 0) * S_LEN + t] = (_Float16)(x.x * sc);
      Vt[(d0 + 1) * S_LEN + t] = (_Float16)(x.y * sc);
      Vt[(d0 + 2) * S_LEN + t] = (_Float16)(x.z * sc);
      Vt[(d0 + 3) * S_LEN + t] = (_Float16)(x.w * sc);
    }
  }
  __syncthreads();

  // ------------- Phase 2: scores = Q K^T via WMMA (64 tiles / 8 waves) -----
  {
    const float* eqs = (const float*)(smem + OFF_EQ);
    const float* eks = (const float*)(smem + OFF_EK);
    for (int tIdx = wave * 8; tIdx < wave * 8 + 8; ++tIdx) {
      const int rb = tIdx >> 5;         // query row-block (0..1)
      const int ct = tIdx & 31;         // key col-tile   (0..31)
      const _Float16* Arow = Qs + (rb * 16 + l16) * D_DIM;
      const _Float16* Brow = Ks + (ct * 16 + l16) * D_DIM;
      v8f c = {0.f, 0.f, 0.f, 0.f, 0.f, 0.f, 0.f, 0.f};
#pragma unroll
      for (int kk = 0; kk < D_DIM; kk += 32) {
        v16h a  = fragA(Arow, kk, hi);
        v16h bm = fragB(Brow, kk, hi);
        c = wmma_f16(a, bm, c);
      }
      const int   n   = ct * 16 + l16;
      const float scn = eks[n];
#pragma unroll
      for (int i = 0; i < 8; ++i) {
        const int m = rb * 16 + hi * 8 + i;          // C layout: lanes>=16 hold M+8
        Sc[m * S_LEN + n] = c[i] * eqs[m] * scn;     // * 2^(eq+ek-14)
      }
    }
  }
  __syncthreads();

  // -- Phase 3: per-row quantize -> softmax(mantissas) -> quantize probs ----
  {
    const int row = tid >> 3;           // 32 rows x 8 threads
    const int sub = tid & 7;            // 64 columns each
    float* S = Sc + row * S_LEN + sub * 64;
    float amax = 0.f;
    for (int j = 0; j < 64; ++j) amax = fmaxf(amax, fabsf(S[j]));
    amax = gmax8(amax);
    float e = floorf(log2f(fmaxf(amax, FLT_MIN_POS)));
    e = fminf(fmaxf(e, -8.f), 7.f);
    const float qs = exp2f(7.f - e);
    float mmax = -1e30f;
    for (int j = 0; j < 64; ++j) {
      float m = rintf(S[j] * qs);                    // round-half-even, like jnp.round
      m = fminf(fmaxf(m, -128.f), 127.f);
      S[j] = m;
      mmax = fmaxf(mmax, m);
    }
    mmax = gmax8(mmax);
    float sum = 0.f;
    for (int j = 0; j < 64; ++j) sum += __expf(S[j] - mmax);
    sum = gsum8(sum);
    const float inv = 1.f / sum;                     // p_max == 1/sum exactly
    float ep = floorf(log2f(fmaxf(inv, FLT_MIN_POS)));
    ep = fminf(fmaxf(ep, -8.f), 7.f);
    const float qp = exp2f(7.f - ep);
    _Float16* P = Ps + row * S_LEN + sub * 64;
    for (int j = 0; j < 64; ++j) {
      const float p  = __expf(S[j] - mmax) * inv;
      const float mp = fminf(fmaxf(rintf(p * qp), -128.f), 127.f);
      P[j] = (_Float16)mp;
    }
    if (sub == 0) ((float*)(smem + OFF_PSC))[row] = exp2f(ep - 7.f);
  }
  __syncthreads();

  // ------------- Phase 4: ctx = P V via WMMA (8 tiles, 1 per wave) ---------
  {
    const float* pscale = (const float*)(smem + OFF_PSC);
    float* Ctx = Sc;                    // reuse first 8KB of score buffer
    const int rb = wave >> 2;           // row-block (0..1)
    const int ct = wave & 3;            // d col-tile (0..3)
    const _Float16* Arow = Ps + (rb * 16 + l16) * S_LEN;
    const _Float16* Brow = Vt + (ct * 16 + l16) * S_LEN;
    v8f c = {0.f, 0.f, 0.f, 0.f, 0.f, 0.f, 0.f, 0.f};
#pragma unroll 4
    for (int kk = 0; kk < S_LEN; kk += 32) {
      v16h a  = fragA(Arow, kk, hi);
      v16h bm = fragB(Brow, kk, hi);
      c = wmma_f16(a, bm, c);
    }
    const int n = ct * 16 + l16;
#pragma unroll
    for (int i = 0; i < 8; ++i) {
      const int m = rb * 16 + hi * 8 + i;
      Ctx[m * D_DIM + n] = c[i] * pscale[m];
    }
  }
  __syncthreads();

  // -- Phase 5: per-row output quantize + dequant (m_c * 2^(e_c-7)) ---------
  {
    const int row = tid >> 3;
    const int sub = tid & 7;
    const float* C = Sc + row * D_DIM + sub * 8;
    float amax = 0.f;
    for (int j = 0; j < 8; ++j) amax = fmaxf(amax, fabsf(C[j]));
    amax = gmax8(amax);
    float e = floorf(log2f(fmaxf(amax, FLT_MIN_POS)));
    e = fminf(fmaxf(e, -8.f), 7.f);
    const float q  = exp2f(7.f - e);
    const float dq = exp2f(e - 7.f);
    float* op = out + ((size_t)b * S_LEN + q0 + row) * D_DIM + sub * 8;
    for (int j = 0; j < 8; ++j) {
      const float m = fminf(fmaxf(rintf(C[j] * q), -128.f), 127.f);
      op[j] = m * dq;
    }
  }
}

extern "C" void kernel_launch(void* const* d_in, const int* in_sizes, int n_in,
                              void* d_out, int out_size, void* d_ws, size_t ws_size,
                              hipStream_t stream) {
  (void)n_in; (void)out_size; (void)d_ws; (void)ws_size;
  const float* mq = (const float*)d_in[0];
  const int*   eq = (const int*)  d_in[1];
  const float* mk = (const float*)d_in[2];
  const int*   ek = (const int*)  d_in[3];
  const float* mv = (const float*)d_in[4];
  const int*   ev = (const int*)  d_in[5];
  float* out = (float*)d_out;

  const int B = in_sizes[0] / (S_LEN * D_DIM);

  // 232KB dynamic LDS (gfx1250 WGP has 320KB); attribute set is idempotent.
  (void)hipFuncSetAttribute((const void*)mxint_attn_head_kernel,
                            hipFuncAttributeMaxDynamicSharedMemorySize,
                            (int)SMEM_BYTES);

  dim3 grid(S_LEN / QT, B, 1);
  dim3 block(NTHREADS, 1, 1);
  mxint_attn_head_kernel<<<grid, block, SMEM_BYTES, stream>>>(mq, eq, mk, ek, mv, ev, out);
}
